// RnnType_18210661335136
// MI455X (gfx1250) — matrix operand order repaired and checked by reading, not compiled, not run on hardware
//
#include <hip/hip_runtime.h>
#include <hip/hip_bf16.h>
#include <math.h>

// ---- problem constants (match reference) ----
#define Bb   128
#define Tt   1024
#define Hh   512
#define INw  128          // C+E = 32+96
#define G4H  2048         // 4*H
#define NBLK 32           // persistent grid: 32 blocks x 256 thr = 256 waves
#define NTHR 256

// LDS layout (elements, bf16). Rows padded: 32B-aligned, stride != 0 mod 64 banks.
#define SROW   656        // 640 (128 Wih | 512 Whh) + 16 pad   -> 1312 B/row
#define SROWP  528        // 512 Wp + 16 pad                    -> 1056 B/row
#define WP_OFF (64 * SROW)                    // after 64 gate-weight rows
#define SMEM_ELEMS (64 * SROW + 16 * SROWP)   // 50432 elems = 100864 B

typedef __attribute__((ext_vector_type(16))) __bf16 bf16x16;
typedef __attribute__((ext_vector_type(8)))  __bf16 bf16x8;
typedef __attribute__((ext_vector_type(8)))  float  f32x8;

// ---------------- WMMA fragment helpers (wave32, 16x16x32 bf16) -------------
// A: 16x32 (MxK): lane%16 -> M; one contiguous 16-elem (32B) run per lane (k = 16*hi + s).
__device__ __forceinline__ bf16x16 load_a_frag(const __bf16* __restrict__ base,
                                               long lda, int m0, int k0, int lane) {
  int m  = m0 + (lane & 15);
  int hi = lane >> 4;
  const __bf16* p = base + (long)m * lda + k0 + hi * 16;
  bf16x8 lo  = *(const bf16x8*)p;
  bf16x8 hi8 = *(const bf16x8*)(p + 8);
  return __builtin_shufflevector(lo, hi8, 0,1,2,3,4,5,6,7,8,9,10,11,12,13,14,15);
}

// B: 32x16 (KxN) with B[k][n] = W[n][k]; W row n lives in LDS at n*stride.
// Per lane one contiguous 16-elem (32B) run -> 2x ds_load_b128.
__device__ __forceinline__ bf16x16 load_b_lds(const __bf16* row_base, int stride,
                                              int k0, int lane) {
  int n  = lane & 15;
  int hi = lane >> 4;
  return *(const bf16x16*)(row_base + n * stride + k0 + hi * 16);
}

__device__ __forceinline__ f32x8 wmma_bf16(bf16x16 a, bf16x16 b, f32x8 c) {
  return __builtin_amdgcn_wmma_f32_16x16x32_bf16(
      false, a, false, b, (short)0, c, false, false);
}

__device__ __forceinline__ float sigf(float x) { return 1.0f / (1.0f + __expf(-x)); }

// ---------------- device-wide barrier (persistent kernel) -------------------
__device__ __forceinline__ void gsync(unsigned* cnt, unsigned* gen) {
  __syncthreads();
  __threadfence();                       // publish this wave's stores (agent scope)
  if (threadIdx.x == 0) {
    unsigned g = __hip_atomic_load(gen, __ATOMIC_RELAXED, __HIP_MEMORY_SCOPE_AGENT);
    if (__hip_atomic_fetch_add(cnt, 1u, __ATOMIC_ACQ_REL, __HIP_MEMORY_SCOPE_AGENT)
        == (unsigned)(NBLK - 1)) {
      __hip_atomic_store(cnt, 0u, __ATOMIC_RELAXED, __HIP_MEMORY_SCOPE_AGENT);
      __hip_atomic_store(gen, g + 1u, __ATOMIC_RELEASE, __HIP_MEMORY_SCOPE_AGENT);
    } else {
      while (__hip_atomic_load(gen, __ATOMIC_ACQUIRE, __HIP_MEMORY_SCOPE_AGENT) == g)
        __builtin_amdgcn_s_sleep(1);
    }
  }
  __syncthreads();
  __threadfence();                       // invalidate local caches before re-reading h
}

// ---------------- LDS weight staging ----------------------------------------
// Block ht stages its 64 gate-weight rows: row r = g*16+j  <->  W row g*512+ht*16+j,
// columns [0,128) from Wih, [128,640) from Whh.
__device__ void stage_gate_weights(const __bf16* __restrict__ Wih,
                                   const __bf16* __restrict__ Whh,
                                   __bf16* lds, int ht) {
#pragma unroll 1
  for (int r = 0; r < 64; ++r) {
    int g = r >> 4, j = r & 15;
    long n = (long)(g * Hh + ht * 16 + j);
    const __bf16* srcI = Wih + n * INw;
    const __bf16* srcH = Whh + n * Hh;
    __bf16* dst = lds + r * SROW;
    if ((int)threadIdx.x < INw) dst[threadIdx.x] = srcI[threadIdx.x];
    for (int k = threadIdx.x; k < Hh; k += NTHR) dst[INw + k] = srcH[k];
  }
}

// Blocks 0..7 stage 16 rows of Wp (their proj n-tile).
__device__ void stage_proj_weights(const __bf16* __restrict__ Wp, __bf16* lds) {
#pragma unroll 1
  for (int j = 0; j < 16; ++j) {
    const __bf16* src = Wp + (long)(blockIdx.x * 16 + j) * Hh;
    __bf16* dst = lds + WP_OFF + j * SROWP;
    for (int k = threadIdx.x; k < Hh; k += NTHR) dst[k] = src[k];
  }
}

// ---------------- fused LSTM step -------------------------------------------
// wave = blockIdx*8 + localwave: m0=(wave&7)*16, h-tile = blockIdx.
// 20 K-chunks x 4 gates = 80 WMMAs; B from LDS; c lives in registers (creg).
__device__ __forceinline__ void lstm_step(int wave, int lane,
    const __bf16* __restrict__ X, long ldx,          // [128 x INw]
    const __bf16* __restrict__ Hin,                  // [128 x 512]
    const __bf16* wlds,                              // LDS gate-weight region
    const float*  __restrict__ bias,                 // [2048]
    float (&creg)[8],                                // register-resident c
    __bf16* __restrict__ Hout) {                     // [128 x 512]
  int m0  = (wave & 7) * 16;
  int hh0 = (wave >> 3) * 16;
  int lz = 0;
  asm volatile("" : "+v"(lz));          // opaque: block LICM across the time loop
  const __bf16* w = wlds + lz;
  f32x8 ai = {}, af = {}, ag = {}, ao = {};
#pragma unroll
  for (int kc = 0; kc < INw / 32; ++kc) {
    bf16x16 a = load_a_frag(X, ldx, m0, kc * 32, lane);
    ai = wmma_bf16(a, load_b_lds(w,             SROW, kc * 32, lane), ai);
    af = wmma_bf16(a, load_b_lds(w + 16 * SROW, SROW, kc * 32, lane), af);
    ag = wmma_bf16(a, load_b_lds(w + 32 * SROW, SROW, kc * 32, lane), ag);
    ao = wmma_bf16(a, load_b_lds(w + 48 * SROW, SROW, kc * 32, lane), ao);
  }
#pragma unroll
  for (int kc = 0; kc < Hh / 32; ++kc) {
    bf16x16 a = load_a_frag(Hin, Hh, m0, kc * 32, lane);
    ai = wmma_bf16(a, load_b_lds(w,             SROW, INw + kc * 32, lane), ai);
    af = wmma_bf16(a, load_b_lds(w + 16 * SROW, SROW, INw + kc * 32, lane), af);
    ag = wmma_bf16(a, load_b_lds(w + 32 * SROW, SROW, INw + kc * 32, lane), ag);
    ao = wmma_bf16(a, load_b_lds(w + 48 * SROW, SROW, INw + kc * 32, lane), ao);
  }
  int hh = hh0 + (lane & 15);
  float bi = bias[hh], bf = bias[Hh + hh], bg = bias[2 * Hh + hh], bo = bias[3 * Hh + hh];
  int row = m0 + ((lane >> 4) << 3);
#pragma unroll
  for (int r = 0; r < 8; ++r) {
    float cn = sigf(af[r] + bf) * creg[r] + sigf(ai[r] + bi) * tanhf(ag[r] + bg);
    creg[r] = cn;
    Hout[(long)(row + r) * Hh + hh] = (__bf16)(sigf(ao[r] + bo) * tanhf(cn));
  }
}

// ---------------- projection: pred = H*Wp^T + bp (blocks 0..7) --------------
__device__ __forceinline__ void proj_step(int wave, int lane,
    const __bf16* __restrict__ Hb,      // [128,512]
    const __bf16* wplds,                // LDS Wp region (this block's 16 rows)
    const float*  __restrict__ bp,      // [128]
    float* __restrict__ out_t,          // out + tflip*IN, row stride T*IN
    __bf16* __restrict__ predbf) {      // [128,128]
  int m0 = (wave & 7) * 16;
  int n0 = (wave >> 3) * 16;
  int lz = 0;
  asm volatile("" : "+v"(lz));
  const __bf16* w = wplds + lz;
  f32x8 acc = {};
#pragma unroll
  for (int kc = 0; kc < Hh / 32; ++kc)
    acc = wmma_bf16(load_a_frag(Hb, Hh, m0, kc * 32, lane),
                    load_b_lds(w, SROWP, kc * 32, lane), acc);
  int row = m0 + ((lane >> 4) << 3);
  int col = n0 + (lane & 15);
  float bv = bp[col];
#pragma unroll
  for (int r = 0; r < 8; ++r) {
    float v = acc[r] + bv;
    out_t[(long)(row + r) * ((long)Tt * INw) + col] = v;
    predbf[(row + r) * INw + col] = (__bf16)v;
  }
}

// ---------------- persistent kernel: whole encoder + decoder ----------------
__global__ __launch_bounds__(NTHR, 1) void k_persistent(
    const __bf16* __restrict__ xbf,
    const __bf16* __restrict__ wihE, const __bf16* __restrict__ whhE,
    const float* __restrict__ bE,
    const __bf16* __restrict__ wihD, const __bf16* __restrict__ whhD,
    const float* __restrict__ bD,
    const __bf16* __restrict__ wp, const float* __restrict__ bp,
    float* __restrict__ out,
    __bf16* hA, __bf16* hB, __bf16* predbf,
    unsigned* cnt, unsigned* gen) {
  extern __shared__ __bf16 smem[];
  int wave = (int)(threadIdx.x >> 5) + (int)blockIdx.x * (NTHR / 32);  // 0..255
  int lane = (int)(threadIdx.x & 31);
  int ht = (int)blockIdx.x;

  // stage encoder weights (+ proj weights for blocks 0..7) into LDS
  stage_gate_weights(wihE, whhE, smem, ht);
  if (blockIdx.x < 8) stage_proj_weights(wp, smem);
  __syncthreads();

  float creg[8];
#pragma unroll
  for (int r = 0; r < 8; ++r) creg[r] = 0.0f;

  const __bf16* hin = hA;   // zeroed by init kernel
  __bf16*       hout = hB;

  // ---- encoder: T-1 steps ----
  for (int t = 0; t < Tt - 1; ++t) {
    lstm_step(wave, lane, xbf + (long)t * INw, (long)Tt * INw,
              hin, smem, bE, creg, hout);
    gsync(cnt, gen);
    const __bf16* tmp = hin; hin = hout; hout = (__bf16*)tmp;
  }

  // ---- swap LDS to decoder weights (block-local; all waves past last gsync) --
  __syncthreads();
  stage_gate_weights(wihD, whhD, smem, ht);
  __syncthreads();

  // ---- pred0 -> out[:, T-1, :] (reference flips time) ----
  if (blockIdx.x < 8)
    proj_step(wave, lane, hin, smem + WP_OFF, bp, out + (long)(Tt - 1) * INw, predbf);
  gsync(cnt, gen);

  // ---- decoder: T-1 autoregressive steps ----
  for (int s = 1; s <= Tt - 1; ++s) {
    lstm_step(wave, lane, predbf, (long)INw,
              hin, smem, bD, creg, hout);
    gsync(cnt, gen);
    const __bf16* tmp = hin; hin = hout; hout = (__bf16*)tmp;
    if (blockIdx.x < 8)
      proj_step(wave, lane, hin, smem + WP_OFF, bp, out + (long)(Tt - 1 - s) * INw, predbf);
    gsync(cnt, gen);
  }
}

// ---------------- setup kernels ---------------------------------------------
__global__ __launch_bounds__(256) void k_concat_x(const float* __restrict__ cont,
                                                  const float* __restrict__ cat,
                                                  __bf16* __restrict__ xbf) {
  long idx = (long)blockIdx.x * 256 + threadIdx.x;   // B*T*IN threads
  int  k   = (int)(idx & (INw - 1));
  long bt  = idx >> 7;
  float v  = (k < 32) ? cont[bt * 32 + k] : cat[bt * 96 + (k - 32)];
  xbf[idx] = (__bf16)v;
}

__global__ __launch_bounds__(256) void k_cvt_bf16(const float* __restrict__ src,
                                                  __bf16* __restrict__ dst, int n) {
  int i = blockIdx.x * 256 + threadIdx.x;
  if (i < n) dst[i] = (__bf16)src[i];
}

__global__ __launch_bounds__(256) void k_init_state(__bf16* __restrict__ hA,
                                                    unsigned* __restrict__ cnt,
                                                    unsigned* __restrict__ gen) {
  int i = blockIdx.x * 256 + threadIdx.x;   // B*H = 65536 threads
  hA[i] = (__bf16)0.0f;
  if (i == 0) { *cnt = 0u; *gen = 0u; }
}

// ---------------- host orchestration ----------------------------------------
extern "C" void kernel_launch(void* const* d_in, const int* in_sizes, int n_in,
                              void* d_out, int out_size, void* d_ws, size_t ws_size,
                              hipStream_t stream) {
  const float* cont   = (const float*)d_in[0];
  const float* cat    = (const float*)d_in[1];
  const float* WihE   = (const float*)d_in[2];
  const float* WhhE   = (const float*)d_in[3];
  const float* bE     = (const float*)d_in[4];
  const float* WihD   = (const float*)d_in[5];
  const float* WhhD   = (const float*)d_in[6];
  const float* bD     = (const float*)d_in[7];
  const float* Wproj  = (const float*)d_in[8];
  const float* bproj  = (const float*)d_in[9];
  float* out = (float*)d_out;
  (void)in_sizes; (void)n_in; (void)out_size; (void)ws_size;

  char* ws = (char*)d_ws;
  size_t off = 0;
  auto alloc = [&](size_t bytes) {
    void* p = ws + off; off = (off + bytes + 255) & ~(size_t)255; return p;
  };
  __bf16*   xbf    = (__bf16*)  alloc((size_t)Bb * Tt * INw * 2);   // 33.5 MB
  __bf16*   wihE   = (__bf16*)  alloc((size_t)G4H * INw * 2);
  __bf16*   whhE   = (__bf16*)  alloc((size_t)G4H * Hh * 2);
  __bf16*   wihD   = (__bf16*)  alloc((size_t)G4H * INw * 2);
  __bf16*   whhD   = (__bf16*)  alloc((size_t)G4H * Hh * 2);
  __bf16*   wp     = (__bf16*)  alloc((size_t)INw * Hh * 2);
  __bf16*   hA     = (__bf16*)  alloc((size_t)Bb * Hh * 2);
  __bf16*   hB     = (__bf16*)  alloc((size_t)Bb * Hh * 2);
  __bf16*   predbf = (__bf16*)  alloc((size_t)Bb * INw * 2);
  unsigned* cnt    = (unsigned*)alloc(256);
  unsigned* gen    = (unsigned*)alloc(256);

  // one-time conversions (parallel, cheap, L2-resident afterwards)
  {
    long n = (long)Bb * Tt * INw;
    k_concat_x<<<(unsigned)(n / 256), 256, 0, stream>>>(cont, cat, xbf);
  }
  k_cvt_bf16<<<(G4H * INw + 255) / 256, 256, 0, stream>>>(WihE, wihE, G4H * INw);
  k_cvt_bf16<<<(G4H * Hh  + 255) / 256, 256, 0, stream>>>(WhhE, whhE, G4H * Hh);
  k_cvt_bf16<<<(G4H * INw + 255) / 256, 256, 0, stream>>>(WihD, wihD, G4H * INw);
  k_cvt_bf16<<<(G4H * Hh  + 255) / 256, 256, 0, stream>>>(WhhD, whhD, G4H * Hh);
  k_cvt_bf16<<<(INw * Hh  + 255) / 256, 256, 0, stream>>>(Wproj, wp, INw * Hh);
  k_init_state<<<(Bb * Hh) / 256, 256, 0, stream>>>(hA, cnt, gen);

  // the whole recurrence in a single persistent kernel (dynamic LDS: ~98.5 KB)
  k_persistent<<<NBLK, NTHR, (unsigned)(SMEM_ELEMS * sizeof(__bf16)), stream>>>(
      xbf, wihE, whhE, bE, wihD, whhD, bD, wp, bproj,
      out, hA, hB, predbf, cnt, gen);
}
